// IntentCVAEGRUModel_14087492731436
// MI455X (gfx1250) — matrix operand order repaired and checked by reading, not compiled, not run on hardware
//
#include <hip/hip_runtime.h>
#include <hip/hip_bf16.h>

// ---------------------------------------------------------------------------
// IntentCVAE GRU decoder for MI455X (gfx1250, wave32, WMMA bf16).
// B=32, T=64, S=128, H=512, V=20000, L=2.
//
// - All GEMMs on v_wmma_f32_16x16x32_bf16 (fp32 accumulate).
// - Weights converted ONCE to fragment-major bf16 (L2-resident, ~31 MB).
// - Activations kept row-major bf16: the ISA A-fragment for 16-bit data is
//   two contiguous 16 B row chunks per lane, so A fragments are loaded
//   directly with two global_load_b128 — no repacking kernels needed.
// - argmax fused into the logits GEMM (shuffle + LDS + atomicMax on packed
//   (ordered_f32, ~col) keys); decoded once after the t-loop.
// ---------------------------------------------------------------------------

constexpr int B = 32;
constexpr int T = 64;
constexpr int S = 128;
constexpr int H = 512;
constexpr int V = 20000;
constexpr int KX = 4 * H;          // 2048 : GRU0 input width
constexpr int NTILES_V = V / 16;   // 1250
constexpr int LOGITS_WAVES = 10;   // 10 waves/block -> 125 blocks, no remainder

typedef __attribute__((ext_vector_type(16))) __bf16 v16bf;
typedef __attribute__((ext_vector_type(8)))  __bf16 v8bf;
typedef __attribute__((ext_vector_type(8)))  float  v8f;

// B-fragment (32x16 KxN tile), fragment-major: 32 lanes x 16 bf16 contiguous.
__device__ __forceinline__ v16bf load_bfrag(const __bf16* base, int fragId, int lane) {
    return *(const v16bf*)(base + ((size_t)fragId << 9) + (lane << 4));
}

// A-fragment (16x32 MxK tile) loaded straight from a row-major bf16 matrix.
// Lane l holds row m; elements e0..7 = K kBase+{0..7}(+8 for upper lanes),
// e8..15 = K kBase+{16..23}(+8) -> two aligned 16 B chunks.
__device__ __forceinline__ v16bf load_a_rm(const __bf16* src, int K, int m,
                                           int kBase, int lane) {
    const __bf16* p = src + (size_t)m * K + kBase + ((lane < 16) ? 0 : 8);
    const v8bf lo = *(const v8bf*)(p);
    const v8bf hi = *(const v8bf*)(p + 16);
    return __builtin_shufflevector(lo, hi, 0, 1, 2, 3, 4, 5, 6, 7,
                                   8, 9, 10, 11, 12, 13, 14, 15);
}

__device__ __forceinline__ v8f wmma_bf16(v16bf a, v16bf b, v8f c) {
    return __builtin_amdgcn_wmma_f32_16x16x32_bf16(false, a, false, b, (short)0, c,
                                                   false, false);
}

// Order-preserving f32 -> u32 map; pack with complemented column so that
// larger packed == (larger value, then smaller column)  (jnp.argmax ties).
__device__ __forceinline__ unsigned long long pack_argmax(float v, int col) {
    unsigned int u = __float_as_uint(v);
    unsigned int ord = (u & 0x80000000u) ? ~u : (u | 0x80000000u);
    return ((unsigned long long)ord << 32) | (unsigned int)(0x7FFFFFFFu - col);
}

// ---------------------------------------------------------------------------
// One-time kernels
// ---------------------------------------------------------------------------
__global__ __launch_bounds__(256)
void init_state_kernel(const float* __restrict__ state,
                       float* __restrict__ h0, float* __restrict__ h1,
                       __bf16* __restrict__ h0b, __bf16* __restrict__ h1b) {
    const int i = blockIdx.x * 256 + threadIdx.x;
    if (i < B * H) {
        const float a = state[i];
        const float b = state[B * H + i];
        h0[i] = a;  h1[i] = b;
        h0b[i] = (__bf16)a;  h1b[i] = (__bf16)b;
    }
}

__global__ __launch_bounds__(128)
void emb_proj_kernel(const float* __restrict__ emotion, const float* __restrict__ intent,
                     const float* __restrict__ eW, const float* __restrict__ iW,
                     float* __restrict__ emo, float* __restrict__ inte) {
    const int b = blockIdx.x;
    for (int h = threadIdx.x; h < H; h += 128) {
        float e = 0.f;
        for (int j = 0; j < 32; ++j) e += emotion[b * 32 + j] * eW[j * H + h];
        float i = 0.f;
        for (int j = 0; j < 9; ++j) i += intent[b * 9 + j] * iW[j * H + h];
        emo[b * H + h] = e;
        inte[b * H + h] = i;
    }
}

// B-fragment builder. src_is_nk==1: src W[N,K], GEMM is X@W^T (B[k][n]=W[n][k]);
// src_is_nk==0: src W[K,N] used directly.
__global__ __launch_bounds__(32)
void make_bfrag_kernel(const float* __restrict__ src, __bf16* __restrict__ dst,
                       int kTiles, int N, int K, int src_is_nk) {
    const int frag = blockIdx.x;
    const int lane = threadIdx.x;
    const int nT = frag / kTiles;
    const int kT = frag % kTiles;
    const int n = nT * 16 + (lane & 15);
    __bf16* out = dst + ((size_t)frag << 9) + (lane << 4);
#pragma unroll
    for (int e = 0; e < 16; ++e) {
        const int k = kT * 32 + (e & 7) + ((e >> 3) << 4) + ((lane < 16) ? 0 : 8);
        float v = 0.f;
        if (n < N && k < K)
            v = src_is_nk ? src[(size_t)n * K + k] : src[(size_t)k * N + n];
        out[e] = (__bf16)v;
    }
}

__global__ __launch_bounds__(256)
void zero_amax_kernel(unsigned long long* __restrict__ amax) {
    const int i = blockIdx.x * 256 + threadIdx.x;
    if (i < T * B) amax[i] = 0ull;
}

// ---------------------------------------------------------------------------
// Per-step: attention + assemble x = [emb[tok], att_ctx, emo, inte] in bf16
// row-major (direct WMMA-A source). One block per batch element (S == 128).
// ---------------------------------------------------------------------------
__global__ __launch_bounds__(128)
void attn_x_kernel(const float* __restrict__ h0, const float* __restrict__ h1,
                   const float* __restrict__ context, const float* __restrict__ mask,
                   const float* __restrict__ emb_w, const int* __restrict__ ys,
                   const int* __restrict__ start_idx,
                   const float* __restrict__ emo, const float* __restrict__ inte,
                   __bf16* __restrict__ xb, int t) {
    __shared__ float s[H];
    __shared__ float sc[S];
    __shared__ float red[S];
    const int b = blockIdx.x;
    const int tid = threadIdx.x;

    for (int h = tid; h < H; h += 128)
        s[h] = 0.5f * (h0[b * H + h] + h1[b * H + h]);
    __syncthreads();

    {   // scores[tid] = context[b, tid, :] . s
        const float* crow = context + ((size_t)b * S + tid) * H;
        float acc = 0.f;
        for (int h = 0; h < H; ++h) acc += crow[h] * s[h];
        sc[tid] = acc;
    }
    __syncthreads();

    red[tid] = sc[tid];
    for (int off = 64; off > 0; off >>= 1) {
        __syncthreads();
        if (tid < off) red[tid] = fmaxf(red[tid], red[tid + off]);
    }
    __syncthreads();
    const float mx = red[0];
    __syncthreads();
    const float ev = __expf(sc[tid] - mx);
    red[tid] = ev;
    for (int off = 64; off > 0; off >>= 1) {
        __syncthreads();
        if (tid < off) red[tid] += red[tid + off];
    }
    __syncthreads();
    const float sum = red[0];
    __syncthreads();
    sc[tid] = (ev / sum) * mask[b * S + tid];   // softmax then mask (ref order)
    __syncthreads();

    const int tok = (t == 0) ? start_idx[0] : ys[b * T + (t - 1)];
    __bf16* xrow = xb + (size_t)b * KX;
    for (int h = tid; h < H; h += 128) {
        float a = 0.f;
        for (int si = 0; si < S; ++si)
            a += sc[si] * context[((size_t)b * S + si) * H + h];
        xrow[h]        = (__bf16)emb_w[(size_t)tok * H + h];
        xrow[512 + h]  = (__bf16)a;
        xrow[1024 + h] = (__bf16)emo[b * H + h];
        xrow[1536 + h] = (__bf16)inte[b * H + h];
    }
}

// ---------------------------------------------------------------------------
// Per-step GRU cell, fully on WMMA. KXT = K tiles of the x-input (64 or 16),
// compile-time so B-fragment addressing strength-reduces to immediates.
//   gi = x @ Wih^T ; gh = h_old @ Whh^T ; gates (r,z,n) -> h_new.
// Writes h_new fp32 in place AND a row-major bf16 mirror (next A operand).
// Grid: 4 blocks x 256 threads; each wave owns a 16-wide h-column strip.
// ---------------------------------------------------------------------------
template <int KXT>
__global__ __launch_bounds__(256)
void gru_kernel(const __bf16* __restrict__ xa,   // [32, KXT*32] row-major bf16
                const __bf16* __restrict__ ha,   // [32, 512]    row-major bf16
                const __bf16* __restrict__ wih, const __bf16* __restrict__ whh,
                const float* __restrict__ b_ih, const float* __restrict__ b_hh,
                float* __restrict__ h_io, __bf16* __restrict__ hb_out) {
    constexpr int KXC = KXT * 32;
    const int tid = threadIdx.x;
    const int wave = tid >> 5;
    const int lane = tid & 31;
    const int cbase = blockIdx.x * 128 + wave * 16;  // h-column strip base
    const int cT = cbase >> 4;                       // tile index in [0,32)
    const int mlo = lane & 15;

    v8f accR[2] = {}, accZ[2] = {}, accIN[2] = {}, accHN[2] = {};

    // gi = x @ Wih^T : gate g lives at gemm columns [g*H + cbase, +16)
    for (int kt = 0; kt < KXT; ++kt) {
        const v16bf a0 = load_a_rm(xa, KXC, mlo, kt * 32, lane);
        const v16bf a1 = load_a_rm(xa, KXC, 16 + mlo, kt * 32, lane);
#pragma unroll
        for (int g = 0; g < 3; ++g) {
            const v16bf bf = load_bfrag(wih, (g * 32 + cT) * KXT + kt, lane);
            v8f* acc = (g == 0) ? accR : (g == 1) ? accZ : accIN;
            acc[0] = wmma_bf16(a0, bf, acc[0]);
            acc[1] = wmma_bf16(a1, bf, acc[1]);
        }
    }
    // gh = h_old @ Whh^T (16 k-tiles); r,z sums share accumulators with gi
    for (int kt = 0; kt < 16; ++kt) {
        const v16bf a0 = load_a_rm(ha, H, mlo, kt * 32, lane);
        const v16bf a1 = load_a_rm(ha, H, 16 + mlo, kt * 32, lane);
#pragma unroll
        for (int g = 0; g < 3; ++g) {
            const v16bf bf = load_bfrag(whh, (g * 32 + cT) * 16 + kt, lane);
            v8f* acc = (g == 0) ? accR : (g == 1) ? accZ : accHN;
            acc[0] = wmma_bf16(a0, bf, acc[0]);
            acc[1] = wmma_bf16(a1, bf, acc[1]);
        }
    }

    // Gates. C layout: lane holds col (lane&15); element e is row
    // mt*16 + e + (lane<16 ? 0 : 8).
    const int col = cbase + mlo;
    const float br  = b_ih[col] + b_hh[col];
    const float bz  = b_ih[H + col] + b_hh[H + col];
    const float bin = b_ih[2 * H + col];
    const float bhn = b_hh[2 * H + col];
#pragma unroll
    for (int mt = 0; mt < 2; ++mt) {
#pragma unroll
        for (int e = 0; e < 8; ++e) {
            const int row = mt * 16 + e + ((lane < 16) ? 0 : 8);
            const float r = 1.f / (1.f + __expf(-(accR[mt][e] + br)));
            const float z = 1.f / (1.f + __expf(-(accZ[mt][e] + bz)));
            const float n = tanhf(accIN[mt][e] + bin + r * (accHN[mt][e] + bhn));
            const float hp = h_io[row * H + col];
            const float hn = (1.f - z) * n + z * hp;
            h_io[row * H + col] = hn;           // fp32 carried state (in place)
            hb_out[row * H + col] = (__bf16)hn; // bf16 mirror for next GEMM
        }
    }
}

// ---------------------------------------------------------------------------
// Per-step: logits = h1 @ w_out + b_out  ([32,512] x [512,20000], WMMA) with
// fused row-argmax. 320 threads = 10 waves x 1 N-tile; grid 125 (exact).
// ---------------------------------------------------------------------------
__global__ __launch_bounds__(320)
void logits_kernel(const __bf16* __restrict__ h1b, const __bf16* __restrict__ wof,
                   const float* __restrict__ b_out, float* __restrict__ logits,
                   unsigned long long* __restrict__ amax, int t) {
    __shared__ unsigned long long sbest[LOGITS_WAVES][B];
    const int lane = threadIdx.x & 31;
    const int wave = threadIdx.x >> 5;
    const int nTile = blockIdx.x * LOGITS_WAVES + wave;
    const int mlo = lane & 15;

    v8f acc0 = {}, acc1 = {};
#pragma unroll
    for (int kt = 0; kt < 16; ++kt) {
        const v16bf a0 = load_a_rm(h1b, H, mlo, kt * 32, lane);
        const v16bf a1 = load_a_rm(h1b, H, 16 + mlo, kt * 32, lane);
        const v16bf bf = load_bfrag(wof, nTile * 16 + kt, lane);
        acc0 = wmma_bf16(a0, bf, acc0);
        acc1 = wmma_bf16(a1, bf, acc1);
    }

    const int col = nTile * 16 + mlo;
    const float bb = b_out[col];
#pragma unroll
    for (int e = 0; e < 8; ++e) {
        const int r0 = e + ((lane < 16) ? 0 : 8);      // batch row in [0,16)
        const float v0 = acc0[e] + bb;
        const float v1 = acc1[e] + bb;
        logits[((size_t)r0 * T + t) * V + col]        = v0;
        logits[((size_t)(r0 + 16) * T + t) * V + col] = v1;

        // per-row max over this tile's 16 columns (held by 16 lanes)
        unsigned long long p0 = pack_argmax(v0, col);
        unsigned long long p1 = pack_argmax(v1, col);
#pragma unroll
        for (int m = 1; m < 16; m <<= 1) {
            const unsigned long long q0 = __shfl_xor(p0, m, 32);
            const unsigned long long q1 = __shfl_xor(p1, m, 32);
            if (q0 > p0) p0 = q0;
            if (q1 > p1) p1 = q1;
        }
        if (mlo == 0) {             // lanes 0 and 16: rows e/e+8 (+16 for acc1)
            sbest[wave][r0] = p0;
            sbest[wave][r0 + 16] = p1;
        }
    }
    __syncthreads();
    if (threadIdx.x < B) {
        unsigned long long m = sbest[0][threadIdx.x];
#pragma unroll
        for (int w = 1; w < LOGITS_WAVES; ++w)
            if (sbest[w][threadIdx.x] > m) m = sbest[w][threadIdx.x];
        atomicMax(&amax[(size_t)t * B + threadIdx.x], m);
    }
}

// Decode all (t, b) argmax keys -> pred, once after the scan.
__global__ __launch_bounds__(256)
void decode_pred_kernel(const unsigned long long* __restrict__ amax,
                        int* __restrict__ pred) {
    const int i = blockIdx.x * 256 + threadIdx.x;
    if (i < T * B) {
        const int t = i / B, b = i % B;
        pred[b * T + t] = (int)(0x7FFFFFFFu - (unsigned int)(amax[i] & 0xFFFFFFFFull));
    }
}

// ---------------------------------------------------------------------------
extern "C" void kernel_launch(void* const* d_in, const int* in_sizes, int n_in,
                              void* d_out, int out_size, void* d_ws, size_t ws_size,
                              hipStream_t stream) {
    (void)in_sizes; (void)n_in; (void)out_size; (void)ws_size;

    const float* emotion   = (const float*)d_in[0];
    const float* intent    = (const float*)d_in[1];
    const float* context   = (const float*)d_in[2];
    const float* state     = (const float*)d_in[3];
    const float* mask      = (const float*)d_in[4];
    const int*   ys        = (const int*)d_in[5];
    const int*   start_idx = (const int*)d_in[6];
    const float* emb_w     = (const float*)d_in[7];
    const float* eEmbW     = (const float*)d_in[8];
    const float* iEmbW     = (const float*)d_in[9];
    const float* w_ih0     = (const float*)d_in[10];
    const float* w_hh0     = (const float*)d_in[11];
    const float* b_ih0     = (const float*)d_in[12];
    const float* b_hh0     = (const float*)d_in[13];
    const float* w_ih1     = (const float*)d_in[14];
    const float* w_hh1     = (const float*)d_in[15];
    const float* b_ih1     = (const float*)d_in[16];
    const float* b_hh1     = (const float*)d_in[17];
    const float* w_out     = (const float*)d_in[18];
    const float* b_out     = (const float*)d_in[19];

    float* logits = (float*)d_out;
    int*   pred   = (int*)(logits + (size_t)B * T * V);

    // ---- workspace carve-out (1 KB aligned; ~31.5 MB total) ----
    char* ws = (char*)d_ws;
    size_t off = 0;
    auto take = [&](size_t bytes) -> char* {
        char* p = ws + off;
        off += (bytes + 1023) & ~(size_t)1023;
        return p;
    };
    float*  emo   = (float*)take((size_t)B * H * 4);
    float*  inte  = (float*)take((size_t)B * H * 4);
    float*  h0    = (float*)take((size_t)B * H * 4);
    float*  h1    = (float*)take((size_t)B * H * 4);
    __bf16* x_bf  = (__bf16*)take((size_t)B * KX * 2);
    __bf16* h0b[2] = {(__bf16*)take((size_t)B * H * 2), (__bf16*)take((size_t)B * H * 2)};
    __bf16* h1b[2] = {(__bf16*)take((size_t)B * H * 2), (__bf16*)take((size_t)B * H * 2)};
    unsigned long long* amax = (unsigned long long*)take((size_t)T * B * 8);
    __bf16* wih0f = (__bf16*)take((size_t)96 * 64 * 512 * 2);
    __bf16* whh0f = (__bf16*)take((size_t)96 * 16 * 512 * 2);
    __bf16* wih1f = (__bf16*)take((size_t)96 * 16 * 512 * 2);
    __bf16* whh1f = (__bf16*)take((size_t)96 * 16 * 512 * 2);
    __bf16* woutf = (__bf16*)take((size_t)NTILES_V * 16 * 512 * 2);

    // ---- pre-pass: state init, projections, weight fragmentization ----
    init_state_kernel<<<(B * H + 255) / 256, 256, 0, stream>>>(state, h0, h1,
                                                               h0b[0], h1b[0]);
    emb_proj_kernel<<<B, 128, 0, stream>>>(emotion, intent, eEmbW, iEmbW, emo, inte);
    zero_amax_kernel<<<(T * B + 255) / 256, 256, 0, stream>>>(amax);
    make_bfrag_kernel<<<96 * 64, 32, 0, stream>>>(w_ih0, wih0f, 64, 3 * H, 4 * H, 1);
    make_bfrag_kernel<<<96 * 16, 32, 0, stream>>>(w_hh0, whh0f, 16, 3 * H, H, 1);
    make_bfrag_kernel<<<96 * 16, 32, 0, stream>>>(w_ih1, wih1f, 16, 3 * H, H, 1);
    make_bfrag_kernel<<<96 * 16, 32, 0, stream>>>(w_hh1, whh1f, 16, 3 * H, H, 1);
    make_bfrag_kernel<<<NTILES_V * 16, 32, 0, stream>>>(w_out, woutf, 16, V, H, 0);

    // ---- sequential decode loop: 4 kernels per step ----
    for (int t = 0; t < T; ++t) {
        const int cur = t & 1, nxt = cur ^ 1;  // ping-pong bf16 state mirrors
        attn_x_kernel<<<B, 128, 0, stream>>>(h0, h1, context, mask, emb_w, ys,
                                             start_idx, emo, inte, x_bf, t);
        gru_kernel<64><<<4, 256, 0, stream>>>(x_bf, h0b[cur],
                                              wih0f, whh0f, b_ih0, b_hh0,
                                              h0, h0b[nxt]);
        gru_kernel<16><<<4, 256, 0, stream>>>(h0b[nxt], h1b[cur],
                                              wih1f, whh1f, b_ih1, b_hh1,
                                              h1, h1b[nxt]);
        logits_kernel<<<NTILES_V / LOGITS_WAVES, 32 * LOGITS_WAVES, 0, stream>>>(
            h1b[nxt], woutf, b_out, logits, amax, t);
    }
    decode_pred_kernel<<<(T * B + 255) / 256, 256, 0, stream>>>(amax, pred);
}
